// GNN_84121229460004
// MI455X (gfx1250) — matrix operand order repaired and checked by reading, not compiled, not run on hardware
//
#include <hip/hip_runtime.h>
#include <hip/hip_bf16.h>
#include <float.h>

// ---------------------------------------------------------------------------
// GATv2 3-layer GNN for MI455X (gfx1250, wave32).
// GEMMs via v_wmma_f32_16x16x32_f16; edge phase via coalesced wave-per-edge
// gathers + f32 global atomics (bandwidth-bound part, ~3 GB -> ~130us floor).
// ---------------------------------------------------------------------------

typedef _Float16 half8 __attribute__((ext_vector_type(8)));
typedef _Float16 v16h  __attribute__((ext_vector_type(16)));
typedef float    v8f   __attribute__((ext_vector_type(8)));

#define N_NODES  20000
#define N_EDGES  320000
#define E_TOT    (N_EDGES + N_NODES)   // with self loops
#define NEG_SLOPE 0.2f

// ---- workspace layout (bytes, every region 256B aligned) ----
static const size_t OFF_A16 = 0;                         // 20000*128 f16
static const size_t OFF_WTL = OFF_A16 + 5120000;         // 256*128 f16 (max)
static const size_t OFF_WTR = OFF_WTL + 65536;
static const size_t OFF_XL  = OFF_WTR + 65536;           // 20000*256 f32
static const size_t OFF_XR  = OFF_XL  + 20480000;
static const size_t OFF_M   = OFF_XR  + 20480000;        // 20000*4 f32
static const size_t OFF_S   = OFF_M   + 320000;
static const size_t OFF_P   = OFF_S   + 320000;          // 340000*4 f32
static const size_t OFF_ACC = OFF_P   + 5440000;         // 20000*256 f32
// total ~72.8 MB

__device__ __forceinline__ void atomicMaxF(float* addr, float val) {
  // standard monotone-bits trick (valid with -FLT_MAX init)
  if (val >= 0.f) atomicMax((int*)addr, __float_as_int(val));
  else            atomicMin((unsigned int*)addr, __float_as_uint(val));
}

// ---------------------------------------------------------------------------
// Dual GEMM: XL = A@Wl + bl ; XR = A@Wr + br.  A:[M,K] f16 row-major,
// B given transposed as Wt:[Npad,K] f16 (K contiguous).  One wave per 16x16
// C-tile; A fragment shared between both products.
// A-frag layout (16-bit A 16x32): lanes0-15 row=lane, elems0..15 ->
// K 0..15; lanes16-31 row=lane-16, K 16..31 (matches B).
// B-frag: lanes0-15 col=lane, K 0..15; lanes16-31 col=lane-16, K 16..31.
// ---------------------------------------------------------------------------
__global__ void gat_gemm_dual(const _Float16* __restrict__ A,
                              const _Float16* __restrict__ BL,
                              const _Float16* __restrict__ BR,
                              const float* __restrict__ biasL,
                              const float* __restrict__ biasR,
                              float* __restrict__ XL,
                              float* __restrict__ XR,
                              int M, int K, int HC, int Npad) {
  const int wavesPerBlk = blockDim.x >> 5;
  int w    = blockIdx.x * wavesPerBlk + (threadIdx.x >> 5);
  int lane = threadIdx.x & 31;
  int tilesN = Npad >> 4;
  int tilesM = M >> 4;               // M is a multiple of 16 (20000)
  if (w >= tilesM * tilesN) return;  // wave-uniform exit, EXEC stays all-1s
  int tm = w / tilesN;
  int tn = w - tm * tilesN;

  int l15  = lane & 15;
  int ksel = lane >> 4;
  const _Float16* arow = A  + (size_t)(tm * 16 + l15) * K;
  const _Float16* blr  = BL + (size_t)(tn * 16 + l15) * K;
  const _Float16* brr  = BR + (size_t)(tn * 16 + l15) * K;

  v8f accL = {};
  v8f accR = {};
  for (int k0 = 0; k0 < K; k0 += 32) {
    half8 alo = *(const half8*)(arow + k0 + ksel * 16);
    half8 ahi = *(const half8*)(arow + k0 + ksel * 16 + 8);
    half8 bl0 = *(const half8*)(blr  + k0 + ksel * 16);
    half8 bl1 = *(const half8*)(blr  + k0 + ksel * 16 + 8);
    half8 br0 = *(const half8*)(brr  + k0 + ksel * 16);
    half8 br1 = *(const half8*)(brr  + k0 + ksel * 16 + 8);
    v16h va, vbl, vbr;
#pragma unroll
    for (int i = 0; i < 8; ++i) {
      va[i]  = alo[i]; va[i + 8]  = ahi[i];
      vbl[i] = bl0[i]; vbl[i + 8] = bl1[i];
      vbr[i] = br0[i]; vbr[i + 8] = br1[i];
    }
    accL = __builtin_amdgcn_wmma_f32_16x16x32_f16(false, va, false, vbl,
                                                  (short)0, accL, false, false);
    accR = __builtin_amdgcn_wmma_f32_16x16x32_f16(false, va, false, vbr,
                                                  (short)0, accR, false, false);
  }
  // C/D layout: VGPR r -> row M = r + 8*ksel, col N = lane&15
  int n = tn * 16 + l15;
  if (n < HC) {
    float bLn = biasL[n];
    float bRn = biasR[n];
#pragma unroll
    for (int r = 0; r < 8; ++r) {
      int m = tm * 16 + r + 8 * ksel;
      XL[(size_t)m * HC + n] = accL[r] + bLn;
      XR[(size_t)m * HC + n] = accR[r] + bRn;
    }
  }
}

// ---------------------------------------------------------------------------
// converts
// ---------------------------------------------------------------------------
__global__ void f32_to_f16k(const float* __restrict__ in,
                            _Float16* __restrict__ out, int n) {
  int t = blockIdx.x * blockDim.x + threadIdx.x;
  if (t < n) out[t] = (_Float16)in[t];
}

// W:[K,HC] f32 -> Wt:[Npad,K] f16 (transposed, zero padded rows)
__global__ void wt_convert(const float* __restrict__ W,
                           _Float16* __restrict__ Wt, int K, int HC, int Npad) {
  int t = blockIdx.x * blockDim.x + threadIdx.x;
  if (t >= Npad * K) return;
  int nrow = t / K;
  int k    = t - nrow * K;
  Wt[t] = (nrow < HC) ? (_Float16)W[(size_t)k * HC + nrow] : (_Float16)0.f;
}

__global__ void gat_init(float* __restrict__ Mx, float* __restrict__ S,
                         float* __restrict__ ACC) {
  int t = blockIdx.x * blockDim.x + threadIdx.x;
  if (t < N_NODES * 4) { Mx[t] = -FLT_MAX; S[t] = 0.f; }
  if (t < N_NODES * 256) ACC[t] = 0.f;
}

// ---------------------------------------------------------------------------
// edge phase, H=4 C=64 (layers 0 and 1): one wave per edge, lane -> 8 channels
// of head lane>>3; butterfly-reduce within the 8-lane head group.
// ---------------------------------------------------------------------------
__device__ __forceinline__ void edge_endpoints(const int* ei, int e, int& s, int& d) {
  if (e < N_EDGES) { s = ei[e]; d = ei[N_EDGES + e]; }
  else             { s = d = e - N_EDGES; }          // self loop
  if ((unsigned)s >= N_NODES) s = 0;
  if ((unsigned)d >= N_NODES) d = 0;
}

__global__ void gat_edge_scores(const int* __restrict__ ei,
                                const float* __restrict__ XL,
                                const float* __restrict__ XR,
                                const float* __restrict__ att,
                                float* __restrict__ Mx,
                                float* __restrict__ P) {
  int w    = (blockIdx.x * blockDim.x + threadIdx.x) >> 5;
  int lane = threadIdx.x & 31;
  if (w >= E_TOT) return;
  int s, d; edge_endpoints(ei, w, s, d);
  int h  = lane >> 3;
  int c0 = (lane & 7) << 3;
  const float4* xs = (const float4*)(XL + (size_t)s * 256 + h * 64 + c0);
  const float4* xd = (const float4*)(XR + (size_t)d * 256 + h * 64 + c0);
  const float4* at = (const float4*)(att + h * 64 + c0);
  float sum = 0.f;
#pragma unroll
  for (int q = 0; q < 2; ++q) {
    float4 a = xs[q], b = xd[q], t = at[q];
    float z;
    z = a.x + b.x; z = z > 0.f ? z : NEG_SLOPE * z; sum += t.x * z;
    z = a.y + b.y; z = z > 0.f ? z : NEG_SLOPE * z; sum += t.y * z;
    z = a.z + b.z; z = z > 0.f ? z : NEG_SLOPE * z; sum += t.z * z;
    z = a.w + b.w; z = z > 0.f ? z : NEG_SLOPE * z; sum += t.w * z;
  }
  sum += __shfl_xor(sum, 1, 32);
  sum += __shfl_xor(sum, 2, 32);
  sum += __shfl_xor(sum, 4, 32);
  if ((lane & 7) == 0) {
    P[(size_t)w * 4 + h] = sum;
    atomicMaxF(&Mx[d * 4 + h], sum);
  }
}

__global__ void gat_edge_exp(const int* __restrict__ ei, float* __restrict__ P,
                             const float* __restrict__ Mx, float* __restrict__ S,
                             int H) {
  int t = blockIdx.x * blockDim.x + threadIdx.x;
  if (t >= E_TOT * H) return;
  int e = t / H, h = t - e * H;
  int s, d; edge_endpoints(ei, e, s, d); (void)s;
  float p = __expf(P[t] - Mx[d * H + h]);
  P[t] = p;
  atomicAdd(&S[d * H + h], p);
}

__global__ void gat_edge_scatter(const int* __restrict__ ei,
                                 const float* __restrict__ XL,
                                 const float* __restrict__ P,
                                 const float* __restrict__ S,
                                 float* __restrict__ ACC) {
  int w    = (blockIdx.x * blockDim.x + threadIdx.x) >> 5;
  int lane = threadIdx.x & 31;
  if (w >= E_TOT) return;
  int s, d; edge_endpoints(ei, w, s, d);
  int h  = lane >> 3;
  int c0 = (lane & 7) << 3;
  float alpha = P[(size_t)w * 4 + h] / (S[d * 4 + h] + 1e-16f);
  const float4* xs = (const float4*)(XL + (size_t)s * 256 + h * 64 + c0);
  float* out = ACC + (size_t)d * 256 + h * 64 + c0;
  float4 v0 = xs[0], v1 = xs[1];
  atomicAdd(out + 0, alpha * v0.x); atomicAdd(out + 1, alpha * v0.y);
  atomicAdd(out + 2, alpha * v0.z); atomicAdd(out + 3, alpha * v0.w);
  atomicAdd(out + 4, alpha * v1.x); atomicAdd(out + 5, alpha * v1.y);
  atomicAdd(out + 6, alpha * v1.z); atomicAdd(out + 7, alpha * v1.w);
}

// ---- scalar edge kernels for layer 2 (H=1, C=1) ----
__global__ void gat_edge_scores_1(const int* __restrict__ ei,
                                  const float* __restrict__ XL,
                                  const float* __restrict__ XR,
                                  const float* __restrict__ att,
                                  float* __restrict__ Mx,
                                  float* __restrict__ P) {
  int e = blockIdx.x * blockDim.x + threadIdx.x;
  if (e >= E_TOT) return;
  int s, d; edge_endpoints(ei, e, s, d);
  float z = XL[s] + XR[d];
  z = z > 0.f ? z : NEG_SLOPE * z;
  float sc = att[0] * z;
  P[e] = sc;
  atomicMaxF(&Mx[d], sc);
}

__global__ void gat_edge_scatter_1(const int* __restrict__ ei,
                                   const float* __restrict__ XL,
                                   const float* __restrict__ P,
                                   const float* __restrict__ S,
                                   float* __restrict__ ACC) {
  int e = blockIdx.x * blockDim.x + threadIdx.x;
  if (e >= E_TOT) return;
  int s, d; edge_endpoints(ei, e, s, d);
  float alpha = P[e] / (S[d] + 1e-16f);
  atomicAdd(&ACC[d], alpha * XL[s]);
}

// ---- finalize ----
// layers 0/1: mean over 4 heads + bias, relu; re-emit f16 as next GEMM's A.
__global__ void gat_finalize_mid(const float* __restrict__ ACC,
                                 const float* __restrict__ bias,
                                 _Float16* __restrict__ A16) {
  int t = blockIdx.x * blockDim.x + threadIdx.x;
  if (t >= N_NODES * 64) return;
  int n = t >> 6, c = t & 63;
  const float* a = ACC + (size_t)n * 256 + c;
  float v = 0.25f * (a[0] + a[64] + a[128] + a[192]) + bias[c];
  v = fmaxf(v, 0.f);
  A16[t] = (_Float16)v;
}

__global__ void gat_finalize_out(const float* __restrict__ ACC,
                                 const float* __restrict__ bias,
                                 float* __restrict__ out) {
  int t = blockIdx.x * blockDim.x + threadIdx.x;
  if (t >= N_NODES) return;
  float v = fmaxf(ACC[t] + bias[0], 0.f);
  out[t] = 1.f / (1.f + __expf(-v));
}

// ---------------------------------------------------------------------------
static inline int cdiv(long a, long b) { return (int)((a + b - 1) / b); }

extern "C" void kernel_launch(void* const* d_in, const int* in_sizes, int n_in,
                              void* d_out, int out_size, void* d_ws, size_t ws_size,
                              hipStream_t stream) {
  (void)in_sizes; (void)n_in; (void)out_size; (void)ws_size;
  const float* x  = (const float*)d_in[0];
  const int*   ei = (const int*)d_in[1];
  const float* Wl[3]   = {(const float*)d_in[2],  (const float*)d_in[8],  (const float*)d_in[14]};
  const float* bl[3]   = {(const float*)d_in[3],  (const float*)d_in[9],  (const float*)d_in[15]};
  const float* Wr[3]   = {(const float*)d_in[4],  (const float*)d_in[10], (const float*)d_in[16]};
  const float* br[3]   = {(const float*)d_in[5],  (const float*)d_in[11], (const float*)d_in[17]};
  const float* att[3]  = {(const float*)d_in[6],  (const float*)d_in[12], (const float*)d_in[18]};
  const float* bias[3] = {(const float*)d_in[7],  (const float*)d_in[13], (const float*)d_in[19]};

  char* ws = (char*)d_ws;
  _Float16* A16 = (_Float16*)(ws + OFF_A16);
  _Float16* WTL = (_Float16*)(ws + OFF_WTL);
  _Float16* WTR = (_Float16*)(ws + OFF_WTR);
  float* XLb = (float*)(ws + OFF_XL);
  float* XRb = (float*)(ws + OFF_XR);
  float* Mb  = (float*)(ws + OFF_M);
  float* Sb  = (float*)(ws + OFF_S);
  float* Pb  = (float*)(ws + OFF_P);
  float* ACC = (float*)(ws + OFF_ACC);

  const int TB = 256;                 // 8 wave32 waves per block
  const int edgeWaveBlocks = cdiv((long)E_TOT * 32, TB);

  // ================= layer 0 (K=128, HC=256) =================
  f32_to_f16k<<<cdiv((long)N_NODES * 128, TB), TB, 0, stream>>>(x, A16, N_NODES * 128);
  wt_convert<<<cdiv(256 * 128, TB), TB, 0, stream>>>(Wl[0], WTL, 128, 256, 256);
  wt_convert<<<cdiv(256 * 128, TB), TB, 0, stream>>>(Wr[0], WTR, 128, 256, 256);
  gat_gemm_dual<<<cdiv((long)(N_NODES / 16) * 16, 8), TB, 0, stream>>>(
      A16, WTL, WTR, bl[0], br[0], XLb, XRb, N_NODES, 128, 256, 256);
  gat_init<<<cdiv((long)N_NODES * 256, TB), TB, 0, stream>>>(Mb, Sb, ACC);
  gat_edge_scores<<<edgeWaveBlocks, TB, 0, stream>>>(ei, XLb, XRb, att[0], Mb, Pb);
  gat_edge_exp<<<cdiv((long)E_TOT * 4, TB), TB, 0, stream>>>(ei, Pb, Mb, Sb, 4);
  gat_edge_scatter<<<edgeWaveBlocks, TB, 0, stream>>>(ei, XLb, Pb, Sb, ACC);
  gat_finalize_mid<<<cdiv((long)N_NODES * 64, TB), TB, 0, stream>>>(ACC, bias[0], A16);

  // ================= layer 1 (K=64, HC=256) =================
  wt_convert<<<cdiv(256 * 64, TB), TB, 0, stream>>>(Wl[1], WTL, 64, 256, 256);
  wt_convert<<<cdiv(256 * 64, TB), TB, 0, stream>>>(Wr[1], WTR, 64, 256, 256);
  gat_gemm_dual<<<cdiv((long)(N_NODES / 16) * 16, 8), TB, 0, stream>>>(
      A16, WTL, WTR, bl[1], br[1], XLb, XRb, N_NODES, 64, 256, 256);
  gat_init<<<cdiv((long)N_NODES * 256, TB), TB, 0, stream>>>(Mb, Sb, ACC);
  gat_edge_scores<<<edgeWaveBlocks, TB, 0, stream>>>(ei, XLb, XRb, att[1], Mb, Pb);
  gat_edge_exp<<<cdiv((long)E_TOT * 4, TB), TB, 0, stream>>>(ei, Pb, Mb, Sb, 4);
  gat_edge_scatter<<<edgeWaveBlocks, TB, 0, stream>>>(ei, XLb, Pb, Sb, ACC);
  gat_finalize_mid<<<cdiv((long)N_NODES * 64, TB), TB, 0, stream>>>(ACC, bias[1], A16);

  // ================= layer 2 (K=64, HC=1, Npad=16) =================
  wt_convert<<<cdiv(16 * 64, TB), TB, 0, stream>>>(Wl[2], WTL, 64, 1, 16);
  wt_convert<<<cdiv(16 * 64, TB), TB, 0, stream>>>(Wr[2], WTR, 64, 1, 16);
  gat_gemm_dual<<<cdiv((long)(N_NODES / 16), 8), TB, 0, stream>>>(
      A16, WTL, WTR, bl[2], br[2], XLb, XRb, N_NODES, 64, 1, 16);
  gat_init<<<cdiv((long)N_NODES * 256, TB), TB, 0, stream>>>(Mb, Sb, ACC);
  gat_edge_scores_1<<<cdiv((long)E_TOT, TB), TB, 0, stream>>>(ei, XLb, XRb, att[2], Mb, Pb);
  gat_edge_exp<<<cdiv((long)E_TOT, TB), TB, 0, stream>>>(ei, Pb, Mb, Sb, 1);
  gat_edge_scatter_1<<<cdiv((long)E_TOT, TB), TB, 0, stream>>>(ei, XLb, Pb, Sb, ACC);
  gat_finalize_out<<<cdiv((long)N_NODES, TB), TB, 0, stream>>>(ACC, bias[2], (float*)d_out);
}
